// SlowAttention_1331439862246
// MI455X (gfx1250) — matrix operand order repaired and checked by reading, not compiled, not run on hardware
//
#include <hip/hip_runtime.h>
#include <hip/hip_bf16.h>
#include <stdint.h>

// Problem constants (match the JAX reference)
#define DIMSZ 2048
#define NTOK  4096

// Block tile: 256 (M) x 128 (N); 8 waves in 4x2, each wave owns 64x64.
#define TM_TILE 256
#define TN_TILE 128
#define TK      32           // K-slab per LDS stage (= WMMA K)
#define LPITCH  40           // LDS row pitch in f16 elems (32 + 8 pad; 80B, 16B-aligned)

typedef __attribute__((ext_vector_type(16))) _Float16 v16h;
typedef __attribute__((ext_vector_type(8)))  float    v8f;
typedef __attribute__((ext_vector_type(4)))  unsigned int u32x4;
typedef __attribute__((ext_vector_type(4)))  int      i32x4;
typedef __attribute__((ext_vector_type(4)))  _Float16 v4h;

union FragH { v16h v; u32x4 u[2]; };

// ---------------------------------------------------------------------------
// CDNA5 async global->LDS staging (ASYNCcnt path), guarded so the build can
// never break: falls back to sync VGPR-roundtrip staging if builtins absent.
// Builtin signature (from hipcc diagnostic): (v4i AS1*, v4i AS3*, Ii, Ii).
// ---------------------------------------------------------------------------
#if __has_builtin(__builtin_amdgcn_global_load_async_to_lds_b128) && \
    __has_builtin(__builtin_amdgcn_s_wait_asynccnt)
#define ASYNC_STAGE 1
typedef __attribute__((address_space(1))) i32x4* as1_i32x4p;
typedef __attribute__((address_space(3))) i32x4* as3_i32x4p;
__device__ __forceinline__ void async_b128(const void* g, void* l) {
  // Global flat addresses are identity; flat LDS pointers carry the LDS
  // offset in the low 32 bits (aperture sits in [63:32]). Integer round-trip
  // avoids illegal address-space casts.
  __builtin_amdgcn_global_load_async_to_lds_b128(
      (as1_i32x4p)(uintptr_t)g,
      (as3_i32x4p)(uintptr_t)(uint32_t)(uintptr_t)l, 0, 0);
}
__device__ __forceinline__ void async_wait0() {
  __builtin_amdgcn_s_wait_asynccnt(0);
}
#else
#define ASYNC_STAGE 0
#endif

// ---------------------------------------------------------------------------
// Staging: A tile 256x32 f16 (1024 x 16B chunks), B tile 128x32 (512 chunks).
// Linear chunk ids so each wave issues 8 rows x 64 contiguous bytes.
// ---------------------------------------------------------------------------
#if ASYNC_STAGE
__device__ __forceinline__ void stage_A_async(const _Float16* __restrict__ g,
                                              int row0, int k0,
                                              unsigned short* lds, int tid) {
#pragma unroll
  for (int i = 0; i < 4; ++i) {
    const int id = tid + 256 * i;
    const int r = id >> 2, c = (id & 3) * 8;
    async_b128(g + (size_t)(row0 + r) * DIMSZ + k0 + c, lds + r * LPITCH + c);
  }
}
__device__ __forceinline__ void stage_B_async(const _Float16* __restrict__ g,
                                              int row0, int k0,
                                              unsigned short* lds, int tid) {
#pragma unroll
  for (int i = 0; i < 2; ++i) {
    const int id = tid + 256 * i;
    const int r = id >> 2, c = (id & 3) * 8;
    async_b128(g + (size_t)(row0 + r) * DIMSZ + k0 + c, lds + r * LPITCH + c);
  }
}
#else
__device__ __forceinline__ void stage_A_sync(const _Float16* __restrict__ g,
                                             int row0, int k0,
                                             unsigned short* lds, int tid) {
#pragma unroll
  for (int i = 0; i < 4; ++i) {
    const int id = tid + 256 * i;
    const int r = id >> 2, c = (id & 3) * 8;
    u32x4 d = *(const u32x4*)(g + (size_t)(row0 + r) * DIMSZ + k0 + c);
    *(u32x4*)(lds + r * LPITCH + c) = d;
  }
}
__device__ __forceinline__ void stage_B_sync(const _Float16* __restrict__ g,
                                             int row0, int k0,
                                             unsigned short* lds, int tid) {
#pragma unroll
  for (int i = 0; i < 2; ++i) {
    const int id = tid + 256 * i;
    const int r = id >> 2, c = (id & 3) * 8;
    u32x4 d = *(const u32x4*)(g + (size_t)(row0 + r) * DIMSZ + k0 + c);
    *(u32x4*)(lds + r * LPITCH + c) = d;
  }
}
#endif

// A-fragment (16x32 f16): lane L -> row M=(L&15); half h=L>>4:
//   elems 0..7 = K 8h..8h+7, elems 8..15 = K 16+8h..16+8h+7  (ISA 7.12.2)
__device__ __forceinline__ FragH load_fragA(const unsigned short* lds,
                                            int row_base, int lane) {
  const int m = lane & 15;
  const int h = lane >> 4;
  const unsigned short* p = lds + (row_base + m) * LPITCH;
  FragH f;
  f.u[0] = *(const u32x4*)(p + 8 * h);
  f.u[1] = *(const u32x4*)(p + 16 + 8 * h);
  return f;
}

// B-fragment (32x16 f16): lane L -> col N=(L&15); half h=L>>4:
//   elems 0..7 = K 8h..8h+7, elems 8..15 = K 16+8h..16+8h+7
__device__ __forceinline__ FragH load_fragB(const unsigned short* lds,
                                            int row_base, int lane) {
  const int n = lane & 15;
  const int h = lane >> 4;
  const unsigned short* p = lds + (row_base + n) * LPITCH;
  FragH f;
  f.u[0] = *(const u32x4*)(p + 8 * h);
  f.u[1] = *(const u32x4*)(p + 16 + 8 * h);
  return f;
}

// One K-step: 4 A-frags + 4 B-frags -> 16 WMMAs (64x64 wave tile)
__device__ __forceinline__ void compute_step(const unsigned short* ldsA,
                                             const unsigned short* ldsB,
                                             int lane, int wm, int wn,
                                             v8f acc[4][4]) {
  FragH a[4], b[4];
#pragma unroll
  for (int i = 0; i < 4; ++i) a[i] = load_fragA(ldsA, wm * 64 + i * 16, lane);
#pragma unroll
  for (int i = 0; i < 4; ++i) b[i] = load_fragB(ldsB, wn * 64 + i * 16, lane);
#pragma unroll
  for (int tm = 0; tm < 4; ++tm)
#pragma unroll
    for (int tn = 0; tn < 4; ++tn)
      acc[tm][tn] = __builtin_amdgcn_wmma_f32_16x16x32_f16(
          false, a[tm].v, false, b[tn].v, (short)0, acc[tm][tn], false, false);
}

// Shared GEMM-NT mainloop: acc += A[rowM0..+256, :] * B[rowN0..+128, :]^T
__device__ __forceinline__ void gemm_core(const _Float16* __restrict__ A,
                                          const _Float16* __restrict__ B,
                                          int rowM0, int rowN0,
                                          unsigned short (*ldsA)[TM_TILE * LPITCH],
                                          unsigned short (*ldsB)[TN_TILE * LPITCH],
                                          int tid, int lane, int wm, int wn,
                                          v8f acc[4][4]) {
#if ASYNC_STAGE
  // Double-buffered: wait -> barrier -> issue next slab -> compute current.
  stage_A_async(A, rowM0, 0, ldsA[0], tid);
  stage_B_async(B, rowN0, 0, ldsB[0], tid);
  int buf = 0;
  for (int k0 = 0; k0 < DIMSZ; k0 += TK, buf ^= 1) {
    async_wait0();
    __syncthreads();
    if (k0 + TK < DIMSZ) {
      stage_A_async(A, rowM0, k0 + TK, ldsA[buf ^ 1], tid);
      stage_B_async(B, rowN0, k0 + TK, ldsB[buf ^ 1], tid);
    }
    compute_step(ldsA[buf], ldsB[buf], lane, wm, wn, acc);
  }
#else
  for (int k0 = 0; k0 < DIMSZ; k0 += TK) {
    stage_A_sync(A, rowM0, k0, ldsA[0], tid);
    stage_B_sync(B, rowN0, k0, ldsB[0], tid);
    __syncthreads();
    compute_step(ldsA[0], ldsB[0], lane, wm, wn, acc);
    __syncthreads();
  }
#endif
}

// ---------------------------------------------------------------------------
// f32 -> f16 conversion, 4 elements/thread
// ---------------------------------------------------------------------------
__global__ __launch_bounds__(256) void cvt_f32_f16_kernel(
    const float* __restrict__ in, _Float16* __restrict__ out, int n4) {
  int i = blockIdx.x * 256 + threadIdx.x;
  if (i < n4) {
    float4 v = ((const float4*)in)[i];
    v4h o;
    o.x = (_Float16)v.x; o.y = (_Float16)v.y;
    o.z = (_Float16)v.z; o.w = (_Float16)v.w;
    ((v4h*)out)[i] = o;
  }
}

// ---------------------------------------------------------------------------
// Fused Q/K projection: out[i,j] = (f16)( sum_d X[i,d]*W[j,d] + bias[j] )
// grid = (DIMSZ/128, NTOK/256, 2), 256 threads
// ---------------------------------------------------------------------------
__global__ __launch_bounds__(256) void proj_qk_kernel(
    const _Float16* __restrict__ X,
    const _Float16* __restrict__ Wq, const _Float16* __restrict__ Wk,
    const float* __restrict__ bq,    const float* __restrict__ bk,
    _Float16* __restrict__ Qo,       _Float16* __restrict__ Ko) {
  __shared__ __align__(16) unsigned short ldsA[2][TM_TILE * LPITCH];
  __shared__ __align__(16) unsigned short ldsB[2][TN_TILE * LPITCH];

  const _Float16* W   = blockIdx.z ? Wk : Wq;
  const float*    bia = blockIdx.z ? bk : bq;
  _Float16*       out = blockIdx.z ? Ko : Qo;

  const int tid  = threadIdx.x;
  const int lane = tid & 31;
  const int w    = tid >> 5;
  const int wm   = w & 3;          // 4 waves along M (64 rows each)
  const int wn   = w >> 2;         // 2 waves along N (64 cols each)
  const int rowM0 = blockIdx.y * TM_TILE;
  const int rowN0 = blockIdx.x * TN_TILE;

  v8f acc[4][4] = {};
  gemm_core(X, W, rowM0, rowN0, ldsA, ldsB, tid, lane, wm, wn, acc);

  // C/D layout: lane L -> N=(L&15); VGPR r -> M = r + 8*(L>>4)
  const int nl = lane & 15;
  const int hh = lane >> 4;
#pragma unroll
  for (int tn = 0; tn < 4; ++tn) {
    const int col = rowN0 + wn * 64 + tn * 16 + nl;
    const float bv = bia[col];
#pragma unroll
    for (int tm = 0; tm < 4; ++tm) {
      const int mrow = rowM0 + wm * 64 + tm * 16 + 8 * hh;
#pragma unroll
      for (int r = 0; r < 8; ++r)
        out[(size_t)(mrow + r) * DIMSZ + col] = (_Float16)(acc[tm][tn][r] + bv);
    }
  }
}

// ---------------------------------------------------------------------------
// scores[i,j] = scale * sum_d Q[i,d] * K[j,d]   (f32 out, NxN)
// grid = (NTOK/128, NTOK/256), 256 threads
// ---------------------------------------------------------------------------
__global__ __launch_bounds__(256) void scores_kernel(
    const _Float16* __restrict__ Q, const _Float16* __restrict__ K,
    float* __restrict__ out, float scale) {
  __shared__ __align__(16) unsigned short ldsA[2][TM_TILE * LPITCH];
  __shared__ __align__(16) unsigned short ldsB[2][TN_TILE * LPITCH];

  const int tid  = threadIdx.x;
  const int lane = tid & 31;
  const int w    = tid >> 5;
  const int wm   = w & 3;
  const int wn   = w >> 2;
  const int rowM0 = blockIdx.y * TM_TILE;
  const int rowN0 = blockIdx.x * TN_TILE;

  v8f acc[4][4] = {};
  gemm_core(Q, K, rowM0, rowN0, ldsA, ldsB, tid, lane, wm, wn, acc);

  const int nl = lane & 15;
  const int hh = lane >> 4;
#pragma unroll
  for (int tn = 0; tn < 4; ++tn) {
    const int col = rowN0 + wn * 64 + tn * 16 + nl;
#pragma unroll
    for (int tm = 0; tm < 4; ++tm) {
      const int mrow = rowM0 + wm * 64 + tm * 16 + 8 * hh;
#pragma unroll
      for (int r = 0; r < 8; ++r)
        out[(size_t)(mrow + r) * NTOK + col] = acc[tm][tn][r] * scale;
    }
  }
}

// ---------------------------------------------------------------------------
extern "C" void kernel_launch(void* const* d_in, const int* in_sizes, int n_in,
                              void* d_out, int out_size, void* d_ws, size_t ws_size,
                              hipStream_t stream) {
  const float* x  = (const float*)d_in[0];
  const float* Wq = (const float*)d_in[1];
  const float* bq = (const float*)d_in[2];
  const float* Wk = (const float*)d_in[3];
  const float* bk = (const float*)d_in[4];
  // d_in[5] (W_v), d_in[6] (b_v): V never reaches the output -> skipped.
  float* out = (float*)d_out;

  // Workspace layout (f16 buffers), 64 MB total
  char* ws = (char*)d_ws;
  _Float16* xh  = (_Float16*)(ws + (size_t) 0 * 1024 * 1024);  // 16 MB [NTOK, DIM]
  _Float16* wqh = (_Float16*)(ws + (size_t)16 * 1024 * 1024);  //  8 MB [DIM, DIM]
  _Float16* wkh = (_Float16*)(ws + (size_t)24 * 1024 * 1024);  //  8 MB
  _Float16* qh  = (_Float16*)(ws + (size_t)32 * 1024 * 1024);  // 16 MB [NTOK, DIM]
  _Float16* kh  = (_Float16*)(ws + (size_t)48 * 1024 * 1024);  // 16 MB

  // 1) downconvert inputs to f16
  {
    int n4x = (NTOK * DIMSZ) / 4;
    int n4w = (DIMSZ * DIMSZ) / 4;
    cvt_f32_f16_kernel<<<n4x / 256, 256, 0, stream>>>(x,  xh,  n4x);
    cvt_f32_f16_kernel<<<n4w / 256, 256, 0, stream>>>(Wq, wqh, n4w);
    cvt_f32_f16_kernel<<<n4w / 256, 256, 0, stream>>>(Wk, wkh, n4w);
  }

  // 2) fused Q and K projections (x @ W^T + b)
  {
    dim3 grid(DIMSZ / TN_TILE, NTOK / TM_TILE, 2);   // (16, 16, 2)
    proj_qk_kernel<<<grid, 256, 0, stream>>>(xh, wqh, wkh, bq, bk, qh, kh);
  }

  // 3) scores = (Q @ K^T) / sqrt(D)
  {
    const float scale = 1.0f / sqrtf((float)DIMSZ);
    dim3 grid(NTOK / TN_TILE, NTOK / TM_TILE);       // (32, 16)
    scores_kernel<<<grid, 256, 0, stream>>>(qh, kh, out, scale);
  }
}
